// STGMFM_71253507441198
// MI455X (gfx1250) — matrix operand order repaired and checked by reading, not compiled, not run on hardware
//
#include <hip/hip_runtime.h>
#include <hip/hip_fp16.h>
#include <math.h>

typedef __attribute__((ext_vector_type(16))) _Float16 v16h;
typedef __attribute__((ext_vector_type(8)))  float    v8f;
typedef __attribute__((ext_vector_type(4)))  int      i32x4;
typedef __attribute__((address_space(1))) i32x4 gi32x4;   // global (AS1) int4
typedef __attribute__((address_space(3))) i32x4 li32x4;   // LDS (AS3) int4

#define B_    256
#define CIN   22
#define WIN_  9
#define SEG_  125
#define OUT_  64
#define KS_   63
#define T_    1125          // WIN_*SEG_
#define PAD_  31            // (KS_-1)/2
#define XROWS 192           // staged rows: 125+62=187, padded
#define NFRAG (KS_*2*4)     // 504 live A-fragments of 512 halves
#define NFRAGP 520          // + zero pad so k+1 preload / k+2 prefetch are unconditional

#if __has_builtin(__builtin_amdgcn_global_load_async_to_lds_b128)
#define USE_ASYNC_STAGE 1
#else
#define USE_ASYNC_STAGE 0
#endif

// ---------------------------------------------------------------------------
// Kernel 1: adjacency normalization, A^2, fused sconv/BN matrix, tbn affine.
// Also writes A_s / A_ta / A_tb blocks of d_out.
// ---------------------------------------------------------------------------
__global__ __launch_bounds__(256) void prep_small(
    const float* __restrict__ ew, const float* __restrict__ adj_a,
    const float* __restrict__ adj_b,
    const float* __restrict__ sconv_w, const float* __restrict__ sbn_g,
    const float* __restrict__ sbn_b, const float* __restrict__ sbn_m,
    const float* __restrict__ sbn_v,
    const float* __restrict__ tbn_g, const float* __restrict__ tbn_b,
    const float* __restrict__ tbn_m, const float* __restrict__ tbn_v,
    float* __restrict__ As2_g, float* __restrict__ Ata2_g,
    float* __restrict__ Atb2_g, float* __restrict__ Wf_g,
    float* __restrict__ biasF_g, float* __restrict__ tsc_g,
    float* __restrict__ tbi_g, float* __restrict__ d_out)
{
  __shared__ float As[484], As2[484], rinv[22];
  __shared__ float At[81], Bt[81], dva[9], dvb[9];
  __shared__ float s2L[64];
  const int tid = threadIdx.x;

  for (int idx = tid; idx < 484; idx += 256) {
    int i = idx / 22, j = idx % 22;
    float v = (i == j) ? 1.f : 0.f;
    if (i > j)      v += ew[i*(i-1)/2 + j];
    else if (j > i) v += ew[j*(j-1)/2 + i];
    As[idx] = v;
  }
  if (tid < 81) {
    int i = tid / 9, j = tid % 9;
    At[tid] = 0.5f*(adj_a[i*9+j] + adj_a[j*9+i]);
    Bt[tid] = 0.5f*(adj_b[i*9+j] + adj_b[j*9+i]);
  }
  __syncthreads();
  if (tid < 22) { float s = 0.f; for (int j = 0; j < 22; ++j) s += As[tid*22+j]; rinv[tid] = rsqrtf(s); }
  if (tid >= 32 && tid < 41) { int i = tid-32; float s = 0.f; for (int j = 0; j < 9; ++j) s += At[i*9+j]; dva[i] = rsqrtf(s); }
  if (tid >= 64 && tid < 73) { int i = tid-64; float s = 0.f; for (int j = 0; j < 9; ++j) s += Bt[i*9+j]; dvb[i] = rsqrtf(s); }
  __syncthreads();
  for (int idx = tid; idx < 484; idx += 256) {
    int i = idx / 22, j = idx % 22;
    As[idx] *= rinv[i]*rinv[j];
  }
  if (tid < 81) {
    int i = tid / 9, j = tid % 9;
    At[tid] *= dva[i]*dva[j];
    Bt[tid] *= dvb[i]*dvb[j];
  }
  __syncthreads();
  for (int idx = tid; idx < 484; idx += 256) {
    int i = idx / 22, j = idx % 22;
    float s = 0.f;
    for (int k = 0; k < 22; ++k) s += As[i*22+k]*As[k*22+j];
    As2[idx] = s; As2_g[idx] = s;
    d_out[1024 + idx] = As[idx];   // A_s   (output #2)
    d_out[1670 + idx] = As[idx];   // A_s   (output #5)
  }
  if (tid < 81) {
    int i = tid / 9, j = tid % 9;
    float sa = 0.f, sb = 0.f;
    for (int k = 0; k < 9; ++k) { sa += At[i*9+k]*At[k*9+j]; sb += Bt[i*9+k]*Bt[k*9+j]; }
    Ata2_g[tid] = sa; Atb2_g[tid] = sb;
    d_out[1508 + tid] = At[tid];   // A_ta
    d_out[1589 + tid] = Bt[tid];   // A_tb
  }
  if (tid < 64) {
    float s2 = sbn_g[tid]*rsqrtf(sbn_v[tid] + 1e-5f);
    s2L[tid] = s2;
    biasF_g[tid] = sbn_b[tid] - s2*sbn_m[tid];
    float ts = tbn_g[tid]*rsqrtf(tbn_v[tid] + 1e-5f);
    tsc_g[tid] = ts;
    tbi_g[tid] = tbn_b[tid] - ts*tbn_m[tid];
  }
  __syncthreads();
  for (int idx = tid; idx < OUT_*CIN; idx += 256) {      // Wf = diag(s2) * Ws * As2
    int o = idx / CIN, j = idx % CIN;
    float acc = 0.f;
    for (int c = 0; c < CIN; ++c) acc += sconv_w[o*CIN + c]*As2[c*22 + j];
    Wf_g[idx] = s2L[o]*acc;
  }
}

// ---------------------------------------------------------------------------
// Kernel 2: swizzle tconv weights (64,64,63) f32 -> f16 WMMA A-fragments,
// in contiguous k order (K = 16*(lane/16) + j; one 32B read per lane per frag).
// Fragments >= NFRAG are zero padding for unconditional k+1/k+2 preloads.
// ---------------------------------------------------------------------------
__global__ __launch_bounds__(256) void prep_wfrag(
    const float* __restrict__ tconv_w, _Float16* __restrict__ Whf)
{
  int idx = blockIdx.x*256 + threadIdx.x;
  if (idx >= NFRAGP*512) return;
  int frag = idx >> 9;
  if (frag >= NFRAG) { Whf[idx] = (_Float16)0.f; return; }
  int j    = idx & 15;
  int lane = (idx >> 4) & 31;
  int m    = frag & 3;
  int kkk  = frag >> 2;
  int kk   = kkk & 1;
  int k    = kkk >> 1;
  int r    = lane & 15;
  int kb   = (lane >> 4)*16;
  int v    = j >> 1, h = j & 1;
  int Kc   = kb + 2*v + h;
  int o    = m*16 + r;
  int i    = kk*32 + Kc;
  Whf[idx] = (_Float16)tconv_w[(o*OUT_ + i)*KS_ + k];
}

// ---------------------------------------------------------------------------
// Kernel 3: fused channel-mix for BOTH branches.
//   yA[b,t,o] = Wf @ x                (spatialGCN^2 + 1x1conv + sbn)
//   yB[b,t,o] = Wf @ (A_tb^2 row-mix of flat x, stride 2750)
// Outputs time-major f16 (B,T,64) ready for WMMA conv staging.
// ---------------------------------------------------------------------------
__global__ __launch_bounds__(256) void stage1(
    const float* __restrict__ x, const float* __restrict__ Wf,
    const float* __restrict__ biasF, const float* __restrict__ Atb2,
    _Float16* __restrict__ yA, _Float16* __restrict__ yB)
{
  __shared__ float XA[CIN*128];
  __shared__ float XB[CIN*128];
  __shared__ float WfL[OUT_*CIN];
  __shared__ float bL[OUT_];
  __shared__ float AtbL[81];
  const int tid = threadIdx.x;
  const int bw = blockIdx.x;
  const int b = bw / WIN_, w = bw % WIN_;
  const float* xb = x + (size_t)b*CIN*T_;

  for (int i = tid; i < OUT_*CIN; i += 256) WfL[i] = Wf[i];
  if (tid < OUT_) bL[tid] = biasF[tid];
  if (tid < 81)   AtbL[tid] = Atb2[tid];
  for (int idx = tid; idx < CIN*SEG_; idx += 256) {
    int j = idx / SEG_, s = idx % SEG_;
    XA[j*128 + s] = xb[j*T_ + w*SEG_ + s];
  }
  __syncthreads();
  for (int idx = tid; idx < CIN*SEG_; idx += 256) {
    int j = idx / SEG_, s = idx % SEG_;
    int f = j*T_ + w*SEG_ + s;            // flat (C,T) index
    int w2 = f / 2750, r = f % 2750;      // branch-B reshape rows
    float acc = 0.f;
    #pragma unroll
    for (int wp = 0; wp < WIN_; ++wp) acc += AtbL[w2*WIN_ + wp]*xb[wp*2750 + r];
    XB[j*128 + s] = acc;
  }
  __syncthreads();
  const int o = tid & 63;
  for (int s = tid >> 6; s < SEG_; s += 4) {
    float aA = bL[o], aB = bL[o];
    #pragma unroll
    for (int j = 0; j < CIN; ++j) {
      float wv = WfL[o*CIN + j];
      aA += wv*XA[j*128 + s];
      aB += wv*XB[j*128 + s];
    }
    size_t off = ((size_t)b*T_ + w*SEG_ + s)*OUT_ + o;
    yA[off] = (_Float16)aA;
    yB[off] = (_Float16)aB;
  }
}

// ---------------------------------------------------------------------------
// Kernel 4: temporal conv (KS=63, 'SAME') as implicit GEMM via WMMA f16,
// fused tbn + exact GELU + per-segment mean pool.
// One block per (batch, window); 8 waves; wave = (M-tile m, 4 N-tiles).
// A-fragments are software-pipelined (k+1 preload + k+2 prefetch); the
// activation tile is staged with ASYNCcnt-tracked global->LDS copies.
// ---------------------------------------------------------------------------
__global__ __launch_bounds__(256) void conv_gelu_pool(
    const _Float16* __restrict__ y,    // (B,T,64) f16, time-major
    const _Float16* __restrict__ Whf,  // NFRAGP fragments * 512 halves
    const float* __restrict__ tscale, const float* __restrict__ tbias,
    float* __restrict__ P)             // (B,64,WIN) pooled
{
  __shared__ _Float16 xs[XROWS*OUT_];  // time-major: xs[tt][ch]
  __shared__ float ps[OUT_];
  __shared__ float tscL[OUT_], tbiL[OUT_];
  const int tid = threadIdx.x;
  const int bw = blockIdx.x;
  const int b = bw / WIN_, w = bw % WIN_;
  const int t0 = w*SEG_ - PAD_;
  const _Float16* yb = y + (size_t)b*T_*OUT_;
  const int NCHUNK = (XROWS*OUT_)/8;   // 16-byte chunks

  if (tid < OUT_) { ps[tid] = 0.f; tscL[tid] = tscale[tid]; tbiL[tid] = tbias[tid]; }

#if USE_ASYNC_STAGE
  // Pre-zero the tile (covers the 'SAME' halo), then async-copy valid rows.
  {
    uint4 z = {0u, 0u, 0u, 0u};
    for (int c = tid; c < NCHUNK; c += 256) *(uint4*)&xs[c << 3] = z;
  }
  __syncthreads();
  for (int c = tid; c < NCHUNK; c += 256) {
    int tt  = c >> 3;
    int col = (c & 7) << 3;
    int t   = t0 + tt;
    if (t >= 0 && t < T_) {
      __builtin_amdgcn_global_load_async_to_lds_b128(
          (gi32x4*)(yb + (size_t)t*OUT_ + col),
          (li32x4*)&xs[tt*OUT_ + col],
          0, 0);
    }
  }
#if __has_builtin(__builtin_amdgcn_s_wait_asynccnt)
  __builtin_amdgcn_s_wait_asynccnt(0);
#else
  asm volatile("s_wait_asynccnt 0" ::: "memory");
#endif
  __syncthreads();
#else
  for (int c = tid; c < NCHUNK; c += 256) {
    int tt  = c >> 3;
    int col = (c & 7) << 3;
    int t   = t0 + tt;
    uint4 v = {0u, 0u, 0u, 0u};
    if (t >= 0 && t < T_) v = *(const uint4*)(yb + (size_t)t*OUT_ + col);
    *(uint4*)&xs[tt*OUT_ + col] = v;
  }
  __syncthreads();
#endif

  const int lane  = tid & 31;
  const int wid   = tid >> 5;
  const int m     = wid & 3;          // M tile (out channels m*16..)
  const int nb    = (wid >> 2)*4;     // first of 4 N tiles (time)
  const int rowl  = lane & 15;
  const int hi    = lane >> 4;
  const int chsel = hi << 4;          // B-frag: lanes>=16 hold K=16..31

  v8f acc0 = {}; v8f acc1 = {}; v8f acc2 = {}; v8f acc3 = {};

  // A-fragment base for this wave; k-stride = 8 frags (4096 halves),
  // kk-stride = 4 frags (2048 halves).
  const _Float16* apb = Whf + (m << 9) + (lane << 4);
  v16h a0 = *(const v16h*)(apb);
  v16h a1 = *(const v16h*)(apb + 2048);

  for (int k = 0; k < KS_; ++k) {
    const _Float16* nap = apb + (size_t)(k + 1)*4096;
    v16h na0 = *(const v16h*)(nap);            // k+1 preload (zero pad at k=62)
    v16h na1 = *(const v16h*)(nap + 2048);
    __builtin_prefetch(nap + 4096, 0, 1);      // k+2 warm (global_prefetch_b8)

    const int tb = rowl + k;
    const int ch0 = chsel, ch1 = 32 + chsel;
    v16h b00 = *(const v16h*)&xs[((nb+0)*16 + tb)*OUT_ + ch0];
    v16h b01 = *(const v16h*)&xs[((nb+1)*16 + tb)*OUT_ + ch0];
    v16h b02 = *(const v16h*)&xs[((nb+2)*16 + tb)*OUT_ + ch0];
    v16h b03 = *(const v16h*)&xs[((nb+3)*16 + tb)*OUT_ + ch0];
    acc0 = __builtin_amdgcn_wmma_f32_16x16x32_f16(false, a0, false, b00, (short)0, acc0, false, false);
    acc1 = __builtin_amdgcn_wmma_f32_16x16x32_f16(false, a0, false, b01, (short)0, acc1, false, false);
    acc2 = __builtin_amdgcn_wmma_f32_16x16x32_f16(false, a0, false, b02, (short)0, acc2, false, false);
    acc3 = __builtin_amdgcn_wmma_f32_16x16x32_f16(false, a0, false, b03, (short)0, acc3, false, false);
    v16h b10 = *(const v16h*)&xs[((nb+0)*16 + tb)*OUT_ + ch1];
    v16h b11 = *(const v16h*)&xs[((nb+1)*16 + tb)*OUT_ + ch1];
    v16h b12 = *(const v16h*)&xs[((nb+2)*16 + tb)*OUT_ + ch1];
    v16h b13 = *(const v16h*)&xs[((nb+3)*16 + tb)*OUT_ + ch1];
    acc0 = __builtin_amdgcn_wmma_f32_16x16x32_f16(false, a1, false, b10, (short)0, acc0, false, false);
    acc1 = __builtin_amdgcn_wmma_f32_16x16x32_f16(false, a1, false, b11, (short)0, acc1, false, false);
    acc2 = __builtin_amdgcn_wmma_f32_16x16x32_f16(false, a1, false, b12, (short)0, acc2, false, false);
    acc3 = __builtin_amdgcn_wmma_f32_16x16x32_f16(false, a1, false, b13, (short)0, acc3, false, false);
    a0 = na0; a1 = na1;
  }

  // Epilogue: tbn affine + exact GELU, fold 4 N-tiles in-register, then
  // 16-lane shfl-xor tree (lanes share the same out channel) -> ds_add_f32.
  const int rowadd = hi*8;   // C/D layout: lanes>=16 hold M = r+8
  #pragma unroll
  for (int r = 0; r < 8; ++r) {
    const int o = m*16 + r + rowadd;
    const float ts = tscL[o], tb2 = tbiL[o];
    float g = 0.f;
    #pragma unroll
    for (int n = 0; n < 4; ++n) {
      float av = (n == 0) ? acc0[r] : ((n == 1) ? acc1[r] : ((n == 2) ? acc2[r] : acc3[r]));
      int s = (nb + n)*16 + rowl;
      float v = av*ts + tb2;
      float e = 0.5f*v*(1.0f + erff(v*0.70710678118654752f));
      g += (s < SEG_) ? e : 0.f;
    }
    g += __shfl_xor(g, 8, 16);
    g += __shfl_xor(g, 4, 16);
    g += __shfl_xor(g, 2, 16);
    g += __shfl_xor(g, 1, 16);
    if (rowl == 0) atomicAdd(&ps[o], g);
  }
  __syncthreads();
  if (tid < OUT_) P[((size_t)b*OUT_ + tid)*WIN_ + w] = ps[tid]*(1.0f/SEG_);
}

// ---------------------------------------------------------------------------
// Kernel 5: tails. P2A = A_ta^2 mix of pooled A; three FCs; fuse -> logits.
// ---------------------------------------------------------------------------
__global__ __launch_bounds__(256) void tail(
    const float* __restrict__ x,
    const float* __restrict__ PA, const float* __restrict__ PB,
    const float* __restrict__ Ata2,
    const float* __restrict__ fc_a_w, const float* __restrict__ fc_a_b,
    const float* __restrict__ fc_b_w, const float* __restrict__ fc_b_b,
    const float* __restrict__ fc_c_w, const float* __restrict__ fc_c_b,
    const float* __restrict__ fuse_w, const float* __restrict__ fuse_b,
    float* __restrict__ d_out)
{
  __shared__ float mc[22];
  __shared__ float p2a[576];
  __shared__ float pbl[576];
  __shared__ float lg[12];
  const int tid = threadIdx.x;
  const int b = blockIdx.x;
  if (tid < 22) mc[tid] = 0.f;
  __syncthreads();
  const float* xb = x + (size_t)b*CIN*T_;
  for (int idx = tid; idx < CIN*T_; idx += 256)
    atomicAdd(&mc[idx / T_], xb[idx]);
  for (int idx = tid; idx < 576; idx += 256)
    pbl[idx] = PB[(size_t)b*576 + idx];
  __syncthreads();
  for (int idx = tid; idx < 576; idx += 256) {
    int o = idx / 9, w = idx % 9;
    float s = 0.f;
    #pragma unroll
    for (int wp = 0; wp < 9; ++wp) s += Ata2[w*9 + wp]*PA[(size_t)b*576 + o*9 + wp];
    p2a[idx] = s;
  }
  __syncthreads();
  if (tid < 4) {
    float a = fc_a_b[tid];
    for (int q = 0; q < 576; ++q) a += fc_a_w[tid*576 + q]*p2a[q];
    lg[tid] = a;
  } else if (tid < 8) {
    int c = tid - 4;
    float a = fc_b_b[c];
    for (int q = 0; q < 576; ++q) a += fc_b_w[c*576 + q]*pbl[q];
    lg[tid] = a;
  } else if (tid < 12) {
    int c = tid - 8;
    float a = fc_c_b[c];
    for (int q = 0; q < 22; ++q) a += fc_c_w[c*22 + q]*(mc[q]*(1.0f/T_));
    lg[tid] = a;
  }
  __syncthreads();
  if (tid < 4) {
    float a = fuse_b[tid];
    #pragma unroll
    for (int q = 0; q < 12; ++q) a += fuse_w[tid*12 + q]*lg[q];
    d_out[b*4 + tid] = a;
  }
}

// ---------------------------------------------------------------------------
extern "C" void kernel_launch(void* const* d_in, const int* in_sizes, int n_in,
                              void* d_out, int out_size, void* d_ws, size_t ws_size,
                              hipStream_t stream) {
  (void)in_sizes; (void)n_in; (void)out_size; (void)ws_size;
  const float* x       = (const float*)d_in[0];
  const float* ew      = (const float*)d_in[1];
  const float* sconv_w = (const float*)d_in[2];
  const float* sbn_g   = (const float*)d_in[3];
  const float* sbn_b   = (const float*)d_in[4];
  const float* sbn_m   = (const float*)d_in[5];
  const float* sbn_v   = (const float*)d_in[6];
  const float* tconv_w = (const float*)d_in[7];
  const float* tbn_g   = (const float*)d_in[8];
  const float* tbn_b   = (const float*)d_in[9];
  const float* tbn_m   = (const float*)d_in[10];
  const float* tbn_v   = (const float*)d_in[11];
  const float* adj_a   = (const float*)d_in[12];
  const float* adj_b   = (const float*)d_in[13];
  const float* fc_a_w  = (const float*)d_in[14];
  const float* fc_a_b  = (const float*)d_in[15];
  const float* fc_b_w  = (const float*)d_in[16];
  const float* fc_b_b  = (const float*)d_in[17];
  const float* fc_c_w  = (const float*)d_in[18];
  const float* fc_c_b  = (const float*)d_in[19];
  const float* fuse_w  = (const float*)d_in[20];
  const float* fuse_b  = (const float*)d_in[21];
  float* out = (float*)d_out;

  char* ws = (char*)d_ws;
  size_t cur = 0;
  auto alloc = [&](size_t bytes) -> char* {
    char* p = ws + cur;
    cur += (bytes + 255) & ~((size_t)255);
    return p;
  };
  float*    As2   = (float*)alloc(484*sizeof(float));
  float*    Ata2  = (float*)alloc(81*sizeof(float));
  float*    Atb2  = (float*)alloc(81*sizeof(float));
  float*    Wf    = (float*)alloc(OUT_*CIN*sizeof(float));
  float*    biasF = (float*)alloc(OUT_*sizeof(float));
  float*    tsc   = (float*)alloc(OUT_*sizeof(float));
  float*    tbi   = (float*)alloc(OUT_*sizeof(float));
  _Float16* Whf   = (_Float16*)alloc((size_t)NFRAGP*512*sizeof(_Float16));
  _Float16* yA    = (_Float16*)alloc((size_t)B_*T_*OUT_*sizeof(_Float16));
  _Float16* yB    = (_Float16*)alloc((size_t)B_*T_*OUT_*sizeof(_Float16));
  float*    PA    = (float*)alloc((size_t)B_*OUT_*WIN_*sizeof(float));
  float*    PB    = (float*)alloc((size_t)B_*OUT_*WIN_*sizeof(float));

  prep_small<<<1, 256, 0, stream>>>(ew, adj_a, adj_b, sconv_w, sbn_g, sbn_b,
                                    sbn_m, sbn_v, tbn_g, tbn_b, tbn_m, tbn_v,
                                    As2, Ata2, Atb2, Wf, biasF, tsc, tbi, out);
  prep_wfrag<<<(NFRAGP*512)/256, 256, 0, stream>>>(tconv_w, Whf);
  stage1<<<B_*WIN_, 256, 0, stream>>>(x, Wf, biasF, Atb2, yA, yB);
  conv_gelu_pool<<<B_*WIN_, 256, 0, stream>>>(yA, Whf, tsc, tbi, PA);
  conv_gelu_pool<<<B_*WIN_, 256, 0, stream>>>(yB, Whf, tsc, tbi, PB);
  tail<<<B_, 256, 0, stream>>>(x, PA, PB, Ata2, fc_a_w, fc_a_b, fc_b_w, fc_b_b,
                               fc_c_w, fc_c_b, fuse_w, fuse_b, out);
}
